// GcnLstmModel_52304111730901
// MI455X (gfx1250) — compile-verified, matrix operand order checked by reading, not run on hardware
//
#include <hip/hip_runtime.h>
#include <cstdint>
#include <cstddef>

// ---- problem constants (from reference setup_inputs) ----
#define NN   2000            // nodes
#define BB   2               // batch
#define TT   24              // timesteps
#define HH   64              // hidden
#define GG   (BB*TT)         // 48 graphs per GAT layer
#define ROWS (GG*NN)         // 96000 GEMM rows
#define BN_  (BB*NN)         // 4000 LSTM sequences
#define NTIL (NN*TT)         // 48000

typedef __attribute__((ext_vector_type(16))) _Float16 v16h;
typedef __attribute__((ext_vector_type(8)))  _Float16 v8h;
typedef __attribute__((ext_vector_type(8)))  float    v8f;

#if defined(__gfx1250__) && __has_builtin(__builtin_amdgcn_tensor_load_to_lds)
#define HAVE_TDM 1
#else
#define HAVE_TDM 0
#endif

// ---------- helpers ----------
__device__ __forceinline__ unsigned f32_ord(float v) {
  unsigned b = __float_as_uint(v);
  return (b & 0x80000000u) ? ~b : (b | 0x80000000u);
}
__device__ __forceinline__ float ord_f32(unsigned u) {
  return (u & 0x80000000u) ? __uint_as_float(u & 0x7FFFFFFFu) : __uint_as_float(~u);
}
__device__ __forceinline__ float sigmoidf_(float x){ return 1.0f/(1.0f+expf(-x)); }
__device__ __forceinline__ float gelu_exact(float x){ return 0.5f*x*(1.0f+erff(x*0.7071067811865475f)); }
__device__ __forceinline__ float lrelu(float x){ return x > 0.f ? x : 0.2f*x; }

// A fragment (16x32 f16) per ISA layout: lane<16 -> row m=lane%16, a[0..7] = K(k0+khi*8 ..+7),
// a[8..15] = K(k0+16+khi*8 ..+7). Built from two contiguous 16B loads (union avoids repacking movs).
__device__ __forceinline__ v16h load_afrag(const _Float16* rowPtr, int k0, int khi) {
  union { v16h v; v8h h[2]; } u;
  u.h[0] = *(const v8h*)(rowPtr + k0 + khi*8);
  u.h[1] = *(const v8h*)(rowPtr + k0 + 16 + khi*8);
  return u.v;
}

// B fragment from pre-swizzled (fragment-major) LDS buffer:
// layout [frag = nt*2+ks][lane][16 halves] -> one contiguous 32B load per lane.
__device__ __forceinline__ v16h load_bfrag_sw(const _Float16* lB, int nt, int ks, int lane) {
  return *(const v16h*)(lB + (((nt << 1) + ks) * 32 + lane) * 16);
}

// packed-fragment index -> (k, n) of the logical B[k][n] matrix (layout independent of Ncols)
__device__ __forceinline__ void unpack_frag_kn(int p, int& k, int& n) {
  int j    = p & 15;
  int lane = (p >> 4) & 31;
  int frag = p >> 9;
  int nt = frag >> 1, ks = frag & 1;
  int khi = lane >> 4, lhalf = lane & 15;
  n = nt * 16 + lhalf;
  k = ks * 32 + khi * 8 + j + (j >= 8 ? 8 : 0);
}

#if HAVE_TDM
typedef unsigned __attribute__((ext_vector_type(4))) v4u;
typedef int      __attribute__((ext_vector_type(4))) v4i;
typedef int      __attribute__((ext_vector_type(8))) v8i_;

// Tensor Data Mover: contiguous global -> LDS copy of nbytes (2-byte elements, single-row tile).
__device__ __forceinline__ void tdm_copy_to_lds(void* ldsDst, const void* gsrc, unsigned nbytes) {
  unsigned nelem = nbytes >> 1;                       // count of 2-byte elements
  unsigned lds_off = (unsigned)(uintptr_t)ldsDst;     // low 32 bits of generic addr = LDS offset
  unsigned long long ga = (unsigned long long)(uintptr_t)gsrc;
  v4u g0;
  g0[0] = 1u;                                         // count=1, is_restore=0, gather off
  g0[1] = lds_off;                                    // lds_addr
  g0[2] = (unsigned)(ga & 0xFFFFFFFFu);               // global_addr[31:0]
  g0[3] = (unsigned)((ga >> 32) & 0x01FFFFFFu) | (2u << 30);   // global_addr[56:32], type=2
  v8i_ g1;
  g1[0] = (int)(1u << 16);                            // data_size = 1 (2 bytes)
  g1[1] = (int)((nelem & 0xFFFFu) << 16);             // tensor_dim0[15:0] in bits 63:48
  g1[2] = (int)((nelem >> 16) & 0xFFFFu) | (1 << 16); // tensor_dim0[31:16], tensor_dim1 = 1
  g1[3] = (int)((nelem & 0xFFFFu) << 16);             // tile_dim0 = nelem
  g1[4] = 1;                                          // tile_dim1 = 1, tile_dim2 = 0
  g1[5] = (int)nelem;                                 // tensor_dim0_stride[31:0]
  g1[6] = 0;
  g1[7] = 0;
  v4i z4 = {0, 0, 0, 0};
#if __clang_major__ >= 23
  v8i_ z8 = {0, 0, 0, 0, 0, 0, 0, 0};
  __builtin_amdgcn_tensor_load_to_lds(g0, g1, z4, z4, z8, 0);
#else
  __builtin_amdgcn_tensor_load_to_lds(g0, g1, z4, z4, 0);
#endif
}
#endif

// ---------- weight prep: f32 -> f16 in WMMA-fragment-swizzled layout ----------
__global__ void prep_kernel(const float* __restrict__ W1, const float* __restrict__ W2,
                            const float* __restrict__ Wih, const float* __restrict__ Whh,
                            const float* __restrict__ bih, const float* __restrict__ bhh,
                            _Float16* __restrict__ W1s, _Float16* __restrict__ W2s,
                            _Float16* __restrict__ WihS, _Float16* __restrict__ WhhS,
                            float* __restrict__ biasz, float* __restrict__ zerob) {
  int i = blockIdx.x * blockDim.x + threadIdx.x;
  if (i < HH*HH) {                       // Ncols=64 weights: B[k][n] = W[k*64+n]
    int k, n; unpack_frag_kn(i, k, n);
    W1s[i] = (_Float16)W1[k*HH + n];
    W2s[i] = (_Float16)W2[k*HH + n];
  }
  if (i < HH*256) {                      // Ncols=256: B[k][n] = W^T[k][n] = W[n*64+k]
    int k, n; unpack_frag_kn(i, k, n);
    WihS[i] = (_Float16)Wih[n*HH + k];
    WhhS[i] = (_Float16)Whh[n*HH + k];
  }
  if (i < 256) { biasz[i] = bih[i] + bhh[i]; zerob[i] = 0.0f; }
}

// ---------- fc: x[b,n,t,8] @ W[8,64] + b, written to graph order (b*T+t, n) as f16 ----------
__global__ void fc_kernel(const float* __restrict__ x, const float* __restrict__ W,
                          const float* __restrict__ b, _Float16* __restrict__ A1) {
  int blk = blockIdx.x;                 // over B*N*T
  int h = threadIdx.x;                  // 64
  int bb = blk / (NN*TT), rem = blk % (NN*TT);
  int n = rem / TT, t = rem % TT;
  const float* xr = x + (size_t)blk * 8;
  float acc = b[h];
#pragma unroll
  for (int k = 0; k < 8; ++k) acc += xr[k] * W[k*HH + h];
  A1[((size_t)(bb*TT + t)*NN + n)*HH + h] = (_Float16)acc;
}

// ---------- generic WMMA GEMM: C[M x Ncols] = A[M x 64](f16) @ Bsw(f16, swizzled) + bias ----------
__global__ void __launch_bounds__(128)
gemm_kernel(const _Float16* __restrict__ A, const _Float16* __restrict__ Bsw,
            const float* __restrict__ bias, float* __restrict__ C, int Ncols) {
  __shared__ __align__(16) _Float16 lB[HH*256];
  int tid = threadIdx.x;
#if HAVE_TDM
  if ((tid >> 5) == 0) {                 // one wave issues the TDM DMA
    tdm_copy_to_lds(lB, Bsw, (unsigned)(HH * Ncols * 2));
    __builtin_amdgcn_s_wait_tensorcnt((short)0);
  }
#else
  {
    int totw = (HH * Ncols) >> 1;
    for (int i = tid; i < totw; i += 128)
      ((unsigned*)lB)[i] = ((const unsigned*)Bsw)[i];
  }
#endif
  __syncthreads();

  int wave = tid >> 5, lane = tid & 31, lhalf = lane & 15, khi = lane >> 4;
  int rowBase = blockIdx.x * 64 + wave * 16;
  const _Float16* arow = A + (size_t)(rowBase + lhalf) * HH;
  v16h a0 = load_afrag(arow, 0, khi);
  v16h a1 = load_afrag(arow, 32, khi);

  for (int nt = 0; nt < (Ncols >> 4); ++nt) {
    int n = nt*16 + lhalf;
    v16h b0 = load_bfrag_sw(lB, nt, 0, lane);
    v16h b1 = load_bfrag_sw(lB, nt, 1, lane);
    v8f c = {};
    c = __builtin_amdgcn_wmma_f32_16x16x32_f16(false, a0, false, b0, (short)0, c, false, false);
    c = __builtin_amdgcn_wmma_f32_16x16x32_f16(false, a1, false, b1, (short)0, c, false, false);
    float bv = bias[n];
#pragma unroll
    for (int r = 0; r < 8; ++r)
      C[(size_t)(rowBase + khi*8 + r) * Ncols + n] = c[r] + bv;
  }
}

// ---------- GAT attention dot products: es = h.asrc, ed = h.adst per (g,n) row ----------
__global__ void dots_kernel(const float* __restrict__ Hg, const float* __restrict__ asrc,
                            const float* __restrict__ adst, float* __restrict__ es,
                            float* __restrict__ ed) {
  int row = blockIdx.x, lane = threadIdx.x;   // blockDim = 32
  const float* hp = Hg + (size_t)row * HH;
  float s = hp[lane]*asrc[lane] + hp[lane+32]*asrc[lane+32];
  float d = hp[lane]*adst[lane] + hp[lane+32]*adst[lane+32];
#pragma unroll
  for (int off = 16; off > 0; off >>= 1) {
    s += __shfl_down(s, off);
    d += __shfl_down(d, off);
  }
  if (lane == 0) { es[row] = s; ed[row] = d; }
}

// ---------- per-layer scratch init: agg=0, max=-inf(encoded), sum=0 ----------
__global__ void init_kernel(unsigned* __restrict__ maxb, float* __restrict__ sumb,
                            float* __restrict__ agg) {
  size_t i = (size_t)blockIdx.x * blockDim.x + threadIdx.x;
  if (i < (size_t)ROWS * HH) agg[i] = 0.0f;
  if (i < (size_t)ROWS) { maxb[i] = 0x007FFFFFu; sumb[i] = 0.0f; }
}

// ---------- edge passes ----------
__global__ void edge_max_kernel(const float* __restrict__ es, const float* __restrict__ ed,
                                const int* __restrict__ src, const int* __restrict__ dst,
                                unsigned* __restrict__ maxb, int E) {
  int tid = blockIdx.x * blockDim.x + threadIdx.x;
  if (tid >= GG * E) return;
  int g = tid / E, e = tid - g * E;
  int s = src[e], d = dst[e];
  float v = lrelu(es[g*NN + s] + ed[g*NN + d]);
  atomicMax(&maxb[g*NN + d], f32_ord(v));
}

__global__ void edge_sum_kernel(const float* __restrict__ es, const float* __restrict__ ed,
                                const int* __restrict__ src, const int* __restrict__ dst,
                                const unsigned* __restrict__ maxb, float* __restrict__ sumb,
                                int E) {
  int tid = blockIdx.x * blockDim.x + threadIdx.x;
  if (tid >= GG * E) return;
  int g = tid / E, e = tid - g * E;
  int s = src[e], d = dst[e];
  int gd = g*NN + d;
  float v = lrelu(es[g*NN + s] + ed[gd]);
  atomicAdd(&sumb[gd], expf(v - ord_f32(maxb[gd])));
}

// one wave per (g, e): 64 channels across 32 lanes
__global__ void edge_msg_kernel(const float* __restrict__ es, const float* __restrict__ ed,
                                const int* __restrict__ src, const int* __restrict__ dst,
                                const unsigned* __restrict__ maxb, const float* __restrict__ sumb,
                                const float* __restrict__ Hg, float* __restrict__ agg, int E) {
  int lane = threadIdx.x & 31;
  int wid = blockIdx.x * (blockDim.x >> 5) + (threadIdx.x >> 5);
  if (wid >= GG * E) return;
  int g = wid / E, e = wid - g * E;
  int s = src[e], d = dst[e];
  int gs = g*NN + s, gd = g*NN + d;
  __builtin_prefetch(Hg + (size_t)gs * HH, 0, 0);
  float v = lrelu(es[gs] + ed[gd]);
  float alpha = expf(v - ord_f32(maxb[gd])) / (sumb[gd] + 1e-16f);
  atomicAdd(&agg[(size_t)gd*HH + lane],      alpha * Hg[(size_t)gs*HH + lane]);
  atomicAdd(&agg[(size_t)gd*HH + lane + 32], alpha * Hg[(size_t)gs*HH + lane + 32]);
}

// ---------- gelu(agg + b); optionally apply the reference's raw-reshape scramble ----------
__global__ void finish_kernel(const float* __restrict__ agg, const float* __restrict__ bvec,
                              _Float16* __restrict__ Aout, int scramble) {
  int q = blockIdx.x, h = threadIdx.x;     // q over ROWS, blockDim 64
  float v = gelu_exact(agg[(size_t)q*HH + h] + bvec[h]);
  int outRow;
  if (scramble) {
    int b = q / NTIL, rem = q % NTIL;
    int n2 = rem / TT, t2 = rem % TT;
    outRow = (b*TT + t2)*NN + n2;
  } else {
    outRow = q;
  }
  Aout[(size_t)outRow*HH + h] = (_Float16)v;
}

// ---------- LSTM: one wave per 16 sequences, WMMA recurrent GEMM per step ----------
__global__ void __launch_bounds__(64)
lstm_kernel(const _Float16* __restrict__ WhhS,   // swizzled [64][256] f16
            const float* __restrict__ xz,        // [BN*T, 256] precomputed x@Wih^T + bias
            float* __restrict__ hfinal) {        // [BN, 64]
  __shared__ __align__(16) _Float16 lW[HH*256];  // 32 KB
  __shared__ __align__(16) _Float16 hT[2][16*HH];
  int tid = threadIdx.x;
#if HAVE_TDM
  if ((tid >> 5) == 0) {
    tdm_copy_to_lds(lW, WhhS, (unsigned)(HH * 256 * 2));
    __builtin_amdgcn_s_wait_tensorcnt((short)0);
  }
#else
  for (int i = tid; i < (HH*256)/2; i += 64)
    ((unsigned*)lW)[i] = ((const unsigned*)WhhS)[i];
#endif
  for (int i = tid; i < (2*16*HH)/2; i += 64)
    ((unsigned*)&hT[0][0])[i] = 0u;
  __syncthreads();

  int wave = tid >> 5, lane = tid & 31, lhalf = lane & 15, khi = lane >> 4;
  int rowBase = blockIdx.x * 32 + wave * 16;
  _Float16* myh = hT[wave];

  float cst[32];
#pragma unroll
  for (int i = 0; i < 32; ++i) cst[i] = 0.0f;

  for (int t = 0; t < TT; ++t) {
    const _Float16* arow = myh + lhalf * HH;
    v16h a0 = load_afrag(arow, 0, khi);
    v16h a1 = load_afrag(arow, 32, khi);
    v8f z[16];
#pragma unroll
    for (int nt = 0; nt < 16; ++nt) {
      int n = nt*16 + lhalf;
#pragma unroll
      for (int r = 0; r < 8; ++r)
        z[nt][r] = xz[((size_t)(rowBase + khi*8 + r)*TT + t)*256 + n];
      v16h b0 = load_bfrag_sw(lW, nt, 0, lane);
      v16h b1 = load_bfrag_sw(lW, nt, 1, lane);
      z[nt] = __builtin_amdgcn_wmma_f32_16x16x32_f16(false, a0, false, b0, (short)0, z[nt], false, false);
      z[nt] = __builtin_amdgcn_wmma_f32_16x16x32_f16(false, a1, false, b1, (short)0, z[nt], false, false);
    }
    __syncthreads();   // all reads of h tiles complete before overwrite
#pragma unroll
    for (int r = 0; r < 8; ++r) {
      int m = khi*8 + r;
#pragma unroll
      for (int q = 0; q < 4; ++q) {
        // lane owns Z[m][hh] for hh = q*16 + lhalf; gates i,f,g,o at col-tiles q, q+4, q+8, q+12
        float iv = z[q][r], fv = z[q+4][r], gv = z[q+8][r], ov = z[q+12][r];
        int ci = r*4 + q;
        float cs = sigmoidf_(fv)*cst[ci] + sigmoidf_(iv)*tanhf(gv);
        cst[ci] = cs;
        float hv = sigmoidf_(ov)*tanhf(cs);
        int hh = q*16 + lhalf;
        myh[m*HH + hh] = (_Float16)hv;
        if (t == TT-1) hfinal[(size_t)(rowBase + m)*HH + hh] = hv;
      }
    }
    __syncthreads();
  }
}

// ---------- head: LayerNorm(last step) * g + b, then dense [64]->[12] ----------
__global__ void head_kernel(const float* __restrict__ hf, const float* __restrict__ lng,
                            const float* __restrict__ lnb, const float* __restrict__ dW,
                            const float* __restrict__ db, float* __restrict__ out) {
  __shared__ float s_n[HH];
  __shared__ float s_r[HH];
  int row = blockIdx.x, t = threadIdx.x;   // blockDim 64
  float v = hf[(size_t)row*HH + t];
  s_r[t] = v; __syncthreads();
#pragma unroll
  for (int off = 32; off > 0; off >>= 1) { if (t < off) s_r[t] += s_r[t+off]; __syncthreads(); }
  float mu = s_r[0] * (1.0f/HH);
  __syncthreads();
  float dv = v - mu;
  s_r[t] = dv*dv; __syncthreads();
#pragma unroll
  for (int off = 32; off > 0; off >>= 1) { if (t < off) s_r[t] += s_r[t+off]; __syncthreads(); }
  float var = s_r[0] * (1.0f/HH);
  float nv = dv * rsqrtf(var + 1e-5f) * lng[t] + lnb[t];
  s_n[t] = nv; __syncthreads();
  if (t < 12) {
    float acc = db[t];
#pragma unroll
    for (int h = 0; h < HH; ++h) acc += s_n[h] * dW[h*12 + t];
    out[(size_t)row*12 + t] = acc;
  }
}

// ---------- launch ----------
extern "C" void kernel_launch(void* const* d_in, const int* in_sizes, int n_in,
                              void* d_out, int out_size, void* d_ws, size_t ws_size,
                              hipStream_t stream) {
  const float* x    = (const float*)d_in[0];
  const int*   esrc = (const int*)d_in[1];
  const int*   edst = (const int*)d_in[2];
  const float* fcW  = (const float*)d_in[3];
  const float* fcb  = (const float*)d_in[4];
  const float* g1W  = (const float*)d_in[5];
  const float* g1b  = (const float*)d_in[6];
  const float* g1as = (const float*)d_in[7];
  const float* g1ad = (const float*)d_in[8];
  const float* g2W  = (const float*)d_in[9];
  const float* g2b  = (const float*)d_in[10];
  const float* g2as = (const float*)d_in[11];
  const float* g2ad = (const float*)d_in[12];
  const float* Wih  = (const float*)d_in[13];
  const float* Whh  = (const float*)d_in[14];
  const float* bih  = (const float*)d_in[15];
  const float* bhh  = (const float*)d_in[16];
  const float* lng  = (const float*)d_in[17];
  const float* lnb  = (const float*)d_in[18];
  const float* dW   = (const float*)d_in[19];
  const float* db   = (const float*)d_in[20];
  const int E = in_sizes[1];

  char* w = (char*)d_ws;
  auto take = [&](size_t bytes) -> char* {
    char* p = w; w += (bytes + 255) & ~(size_t)255; return p;
  };
  _Float16* W1s   = (_Float16*)take(HH*HH*sizeof(_Float16));
  _Float16* W2s   = (_Float16*)take(HH*HH*sizeof(_Float16));
  _Float16* WihS  = (_Float16*)take(HH*256*sizeof(_Float16));
  _Float16* WhhS  = (_Float16*)take(HH*256*sizeof(_Float16));
  float*    biasz = (float*)   take(256*sizeof(float));
  float*    zerob = (float*)   take(256*sizeof(float));
  _Float16* A1    = (_Float16*)take((size_t)ROWS*HH*sizeof(_Float16));
  _Float16* A2    = (_Float16*)take((size_t)ROWS*HH*sizeof(_Float16));
  _Float16* A3    = (_Float16*)take((size_t)ROWS*HH*sizeof(_Float16));
  float*    Hg    = (float*)   take((size_t)ROWS*HH*sizeof(float));
  float*    esb   = (float*)   take((size_t)ROWS*sizeof(float));
  float*    edb   = (float*)   take((size_t)ROWS*sizeof(float));
  unsigned* maxb  = (unsigned*)take((size_t)ROWS*sizeof(unsigned));
  float*    sumb  = (float*)   take((size_t)ROWS*sizeof(float));
  float*    agg   = (float*)   take((size_t)ROWS*HH*sizeof(float));
  float*    xz    = (float*)   take((size_t)ROWS*256*sizeof(float));
  float*    hfin  = (float*)   take((size_t)BN_*HH*sizeof(float));

  const int edgeThreads = GG * E;
  const int edgeBlocks  = (edgeThreads + 255) / 256;
  const int msgBlocks   = (edgeThreads + 7) / 8;     // 8 waves / 256-thread block
  const int initBlocks  = (ROWS*HH + 255) / 256;

  prep_kernel<<<(HH*256 + 255)/256, 256, 0, stream>>>(g1W, g2W, Wih, Whh, bih, bhh,
                                                      W1s, W2s, WihS, WhhS, biasz, zerob);
  fc_kernel<<<BB*NN*TT, HH, 0, stream>>>(x, fcW, fcb, A1);

  // ---- GAT layer 1 ----
  gemm_kernel<<<ROWS/64, 128, 0, stream>>>(A1, W1s, zerob, Hg, HH);
  dots_kernel<<<ROWS, 32, 0, stream>>>(Hg, g1as, g1ad, esb, edb);
  init_kernel<<<initBlocks, 256, 0, stream>>>(maxb, sumb, agg);
  edge_max_kernel<<<edgeBlocks, 256, 0, stream>>>(esb, edb, esrc, edst, maxb, E);
  edge_sum_kernel<<<edgeBlocks, 256, 0, stream>>>(esb, edb, esrc, edst, maxb, sumb, E);
  edge_msg_kernel<<<msgBlocks, 256, 0, stream>>>(esb, edb, esrc, edst, maxb, sumb, Hg, agg, E);
  finish_kernel<<<ROWS, HH, 0, stream>>>(agg, g1b, A2, 1);   // scrambled write for layer 2

  // ---- GAT layer 2 ----
  gemm_kernel<<<ROWS/64, 128, 0, stream>>>(A2, W2s, zerob, Hg, HH);
  dots_kernel<<<ROWS, 32, 0, stream>>>(Hg, g2as, g2ad, esb, edb);
  init_kernel<<<initBlocks, 256, 0, stream>>>(maxb, sumb, agg);
  edge_max_kernel<<<edgeBlocks, 256, 0, stream>>>(esb, edb, esrc, edst, maxb, E);
  edge_sum_kernel<<<edgeBlocks, 256, 0, stream>>>(esb, edb, esrc, edst, maxb, sumb, E);
  edge_msg_kernel<<<msgBlocks, 256, 0, stream>>>(esb, edb, esrc, edst, maxb, sumb, Hg, agg, E);
  finish_kernel<<<ROWS, HH, 0, stream>>>(agg, g2b, A3, 0);   // identity: LSTM reads flat

  // ---- LSTM ----
  gemm_kernel<<<ROWS/64, 128, 0, stream>>>(A3, WihS, biasz, xz, 256);  // x @ Wih^T + (bih+bhh)
  lstm_kernel<<<BN_/32, 64, 0, stream>>>(WhhS, xz, hfin);

  // ---- LayerNorm(last step) + dense head ----
  head_kernel<<<BN_, HH, 0, stream>>>(hfin, lng, lnb, dW, db, (float*)d_out);
}
